// ConvLSTM_36859409334339
// MI455X (gfx1250) — matrix operand order repaired and checked by reading, not compiled, not run on hardware
//
#include <hip/hip_runtime.h>

typedef _Float16 v16h __attribute__((ext_vector_type(16)));
typedef _Float16 v8h  __attribute__((ext_vector_type(8)));
typedef float    v8f  __attribute__((ext_vector_type(8)));

#define NB   4
#define CIN  64
#define CHID 64
#define C4   256
#define HH   64
#define WWID 64
#define HW   4096

// ---------------- utility kernels ----------------
__global__ void zero3_kernel(float* a, int na, float* b, int nb, float* c, int nc) {
  int i = blockIdx.x * blockDim.x + threadIdx.x;
  if (i < na) a[i] = 0.f;
  else if (i < na + nb) b[i - na] = 0.f;
  else if (i < na + nb + nc) c[i - na - nb] = 0.f;
}

// Wx [256][64][3][3] -> f16 [co][k], k = tap*64 + ch  (tap = ky*3+kx)
__global__ void wx_f16_kernel(const float* __restrict__ Wx, _Float16* __restrict__ Wxf) {
  int idx = blockIdx.x * blockDim.x + threadIdx.x;
  if (idx >= C4 * 576) return;
  int co = idx / 576, k = idx % 576;
  int tap = k >> 6, ch = k & 63;
  Wxf[idx] = (_Float16)Wx[co * 576 + ch * 9 + tap];
}

// fold BN into fused weights: Wf [co][k], k = tap*128 + ch (ch<64: scaled Wx, else Wh)
__global__ void wprep_kernel(const float* __restrict__ Wx, const float* __restrict__ Wh,
                             const float* __restrict__ bx, const float* __restrict__ gamma,
                             const float* __restrict__ beta, const float* __restrict__ stats,
                             _Float16* __restrict__ Wf, float* __restrict__ biasc) {
  int idx = blockIdx.x * blockDim.x + threadIdx.x;
  if (idx >= C4 * 1152) return;
  int co = idx / 1152, k = idx % 1152;
  int tap = k >> 7, ch = k & 127;
  const float cnt = 262144.f;  // T*N*H*W
  float mean = stats[co] / cnt;
  float var  = stats[C4 + co] / cnt - mean * mean;
  float sc   = gamma[co] * rsqrtf(var + 1e-5f);
  float v = (ch < 64) ? Wx[co * 576 + ch * 9 + tap] * sc
                      : Wh[co * 576 + (ch - 64) * 9 + tap];
  Wf[idx] = (_Float16)v;
  if (idx < C4) {
    float m2 = stats[idx] / cnt;
    float v2 = stats[C4 + idx] / cnt - m2 * m2;
    biasc[idx] = gamma[idx] * rsqrtf(v2 + 1e-5f) * (bx[idx] - m2) + beta[idx];
  }
}

// ---------------- BN statistics pass: conv_x2h implicit GEMM + atomic reduce ----------------
#define STR_S 72   // LDS channel stride (halves); 72*2=144=9*16 -> 16B aligned b128 reads

__global__ void __launch_bounds__(256)
stats_gemm_kernel(const float* __restrict__ x, const _Float16* __restrict__ Wxf,
                  const float* __restrict__ bx, float* __restrict__ stats) {
  __shared__ __align__(16) _Float16 sm[3 * 66 * STR_S];
  const int tid = threadIdx.x;
  const int bid = blockIdx.x;   // 0..4095 : (frame, row)
  const int f = bid >> 6;       // frame = t*N + n
  const int y = bid & 63;

  // stage 3 rows x 66 cols x 64 ch, zero-padded halo, fp32 -> f16
  for (int i = tid; i < 3 * 66 * 64; i += 256) {
    int ch = i / 198, sp = i % 198;
    int r3 = sp / 66, cc = sp % 66;
    int yy = y + r3 - 1, xx = cc - 1;
    float v = 0.f;
    if (yy >= 0 && yy < HH && xx >= 0 && xx < WWID)
      v = x[(size_t)(f * CIN + ch) * HW + yy * WWID + xx];
    sm[sp * STR_S + ch] = (_Float16)v;
  }
  __syncthreads();

  const int lane = tid & 31, w = tid >> 5;
  const int hg = w & 3, mh = w >> 2;
  const int l15 = lane & 15, kh = lane >> 4;

  v8f acc[2][4];
#pragma unroll
  for (int mt = 0; mt < 2; ++mt)
#pragma unroll
    for (int nt = 0; nt < 4; ++nt)
      acc[mt][nt] = v8f{0.f, 0.f, 0.f, 0.f, 0.f, 0.f, 0.f, 0.f};

  union H16 { v16h v; v8h h[2]; };

  for (int kk = 0; kk < 18; ++kk) {   // K = 576 = 18 x 32
    int tap = kk >> 1, ch0 = (kk & 1) * 32;
    int dy = tap / 3, dx = tap % 3;
    H16 a[2], b[4];
#pragma unroll
    for (int mt = 0; mt < 2; ++mt) {
      int xm = 32 * mh + 16 * mt + l15;
      int base = (dy * 66 + xm + dx) * STR_S + ch0 + kh * 8;
      a[mt].h[0] = *(const v8h*)(sm + base);
      a[mt].h[1] = *(const v8h*)(sm + base + 16);
    }
#pragma unroll
    for (int nt = 0; nt < 4; ++nt) {
      int n = 16 * (hg + 4 * nt) + l15;
      const _Float16* wp = Wxf + n * 576 + kk * 32 + kh * 8;
      b[nt].h[0] = *(const v8h*)(wp);
      b[nt].h[1] = *(const v8h*)(wp + 16);
    }
#pragma unroll
    for (int mt = 0; mt < 2; ++mt)
#pragma unroll
      for (int nt = 0; nt < 4; ++nt)
        acc[mt][nt] = __builtin_amdgcn_wmma_f32_16x16x32_f16(
            false, a[mt].v, false, b[nt].v, (short)0, acc[mt][nt], false, false);
  }

  // per-lane partial sum / sumsq of (conv + bias), one channel per (lane, nt)
#pragma unroll
  for (int nt = 0; nt < 4; ++nt) {
    int nch = 16 * (hg + 4 * nt) + l15;
    float bv = bx[nch];
    float s = 0.f, q = 0.f;
#pragma unroll
    for (int mt = 0; mt < 2; ++mt)
#pragma unroll
      for (int r = 0; r < 8; ++r) {
        float v = acc[mt][nt][r] + bv;
        s += v; q += v * v;
      }
    atomicAdd(&stats[nch], s);
    atomicAdd(&stats[C4 + nch], q);
  }
}

// ---------------- fused recurrent step: [x_t;h] implicit GEMM + LSTM gating ----------------
#define STR_F 136  // 136*2=272=17*16 -> 16B aligned

__global__ void __launch_bounds__(256)
step_kernel(const float* __restrict__ xt, const float* __restrict__ hprev,
            const float* __restrict__ cprev, const _Float16* __restrict__ Wf,
            const float* __restrict__ biasc, float* __restrict__ hnext,
            float* __restrict__ cnext, float* __restrict__ out) {
  __shared__ __align__(16) _Float16 sm[3 * 66 * STR_F];
  const int tid = threadIdx.x;
  const int bid = blockIdx.x;   // 0..255 : (n, y)
  const int n = bid >> 6, y = bid & 63;

  // stage 3 rows x 66 cols x 128 ch  (ch<64 from x_t, else from h_prev)
  for (int i = tid; i < 3 * 66 * 128; i += 256) {
    int ch = i / 198, sp = i % 198;
    int r3 = sp / 66, cc = sp % 66;
    int yy = y + r3 - 1, xx = cc - 1;
    float v = 0.f;
    if (yy >= 0 && yy < HH && xx >= 0 && xx < WWID) {
      if (ch < 64) v = xt[(size_t)(n * CIN + ch) * HW + yy * WWID + xx];
      else         v = hprev[(size_t)(n * CHID + (ch - 64)) * HW + yy * WWID + xx];
    }
    sm[sp * STR_F + ch] = (_Float16)v;
  }
  __syncthreads();

  const int lane = tid & 31, w = tid >> 5;
  const int hg = w & 3, mh = w >> 2;   // wave owns gates {hg, hg+4, hg+8, hg+12} for pixels [32*mh, 32*mh+32)
  const int l15 = lane & 15, kh = lane >> 4;

  v8f acc[2][4];
#pragma unroll
  for (int mt = 0; mt < 2; ++mt)
#pragma unroll
    for (int nt = 0; nt < 4; ++nt)
      acc[mt][nt] = v8f{0.f, 0.f, 0.f, 0.f, 0.f, 0.f, 0.f, 0.f};

  union H16 { v16h v; v8h h[2]; };

  for (int kk = 0; kk < 36; ++kk) {   // K = 1152 = 36 x 32
    int tap = kk >> 2, ch0 = (kk & 3) * 32;
    int dy = tap / 3, dx = tap % 3;
    H16 a[2], b[4];
#pragma unroll
    for (int mt = 0; mt < 2; ++mt) {
      int xm = 32 * mh + 16 * mt + l15;
      int base = (dy * 66 + xm + dx) * STR_F + ch0 + kh * 8;
      a[mt].h[0] = *(const v8h*)(sm + base);
      a[mt].h[1] = *(const v8h*)(sm + base + 16);
    }
#pragma unroll
    for (int nt = 0; nt < 4; ++nt) {
      int nc = 16 * (hg + 4 * nt) + l15;
      const _Float16* wp = Wf + nc * 1152 + kk * 32 + kh * 8;
      b[nt].h[0] = *(const v8h*)(wp);
      b[nt].h[1] = *(const v8h*)(wp + 16);
    }
#pragma unroll
    for (int mt = 0; mt < 2; ++mt)
#pragma unroll
      for (int nt = 0; nt < 4; ++nt)
        acc[mt][nt] = __builtin_amdgcn_wmma_f32_16x16x32_f16(
            false, a[mt].v, false, b[nt].v, (short)0, acc[mt][nt], false, false);
  }

  // LSTM gating, fully register-local: nt=0/1/2/3 -> i/f/o/g for hidden channel hc
  const int hc = 16 * hg + l15;
  const float bi = biasc[hc], bf = biasc[hc + 64], bo = biasc[hc + 128], bg = biasc[hc + 192];
#pragma unroll
  for (int mt = 0; mt < 2; ++mt) {
#pragma unroll
    for (int r = 0; r < 8; ++r) {
      int m = 32 * mh + 16 * mt + 8 * kh + r;   // pixel x within row
      size_t idx = (size_t)(n * CHID + hc) * HW + y * WWID + m;
      float iv = 1.f / (1.f + __expf(-(acc[mt][0][r] + bi)));
      float fv = 1.f / (1.f + __expf(-(acc[mt][1][r] + bf)));
      float ov = 1.f / (1.f + __expf(-(acc[mt][2][r] + bo)));
      float gv = tanhf(acc[mt][3][r] + bg);
      float cv = fv * cprev[idx] + iv * gv;
      cnext[idx] = cv;
      float hv = ov * tanhf(cv);
      hnext[idx] = hv;
      out[idx] = hv;
    }
  }
}

// ---------------- host launcher ----------------
extern "C" void kernel_launch(void* const* d_in, const int* in_sizes, int n_in,
                              void* d_out, int out_size, void* d_ws, size_t ws_size,
                              hipStream_t stream) {
  (void)in_sizes; (void)n_in; (void)out_size; (void)ws_size;
  const float* x     = (const float*)d_in[0];
  const float* Wx    = (const float*)d_in[1];
  const float* bx    = (const float*)d_in[2];
  const float* gamma = (const float*)d_in[3];
  const float* beta  = (const float*)d_in[4];
  const float* Wh    = (const float*)d_in[5];
  float* out = (float*)d_out;

  char* ws = (char*)d_ws;
  _Float16* Wxf  = (_Float16*)(ws);              //  294912 B
  _Float16* Wf   = (_Float16*)(ws + 0x060000);   //  589824 B
  float*    stats= (float*)   (ws + 0x100000);   //    2048 B
  float*    biasc= (float*)   (ws + 0x100800);   //    1024 B
  float*    h0   = (float*)   (ws + 0x200000);   // 4 MB each
  float*    h1   = (float*)   (ws + 0x600000);
  float*    c0   = (float*)   (ws + 0xA00000);
  float*    c1   = (float*)   (ws + 0xE00000);

  const int nz = NB * CHID * HW;  // 1048576
  zero3_kernel<<<(2 * nz + 512 + 255) / 256, 256, 0, stream>>>(h0, nz, c0, nz, stats, 512);
  wx_f16_kernel<<<(C4 * 576 + 255) / 256, 256, 0, stream>>>(Wx, Wxf);
  stats_gemm_kernel<<<4096, 256, 0, stream>>>(x, Wxf, bx, stats);
  wprep_kernel<<<(C4 * 1152 + 255) / 256, 256, 0, stream>>>(Wx, Wh, bx, gamma, beta, stats, Wf, biasc);

  for (int t = 0; t < 16; ++t) {
    const float* hp = (t & 1) ? h1 : h0;
    float*       hn = (t & 1) ? h0 : h1;
    const float* cp = (t & 1) ? c1 : c0;
    float*       cn = (t & 1) ? c0 : c1;
    step_kernel<<<256, 256, 0, stream>>>(x + (size_t)t * nz, hp, cp, Wf, biasc,
                                         hn, cn, out + (size_t)t * nz);
  }
}
